// Graph_Embedding_18348100288742
// MI455X (gfx1250) — matrix operand order repaired and checked by reading, not compiled, not run on hardware
//
#include <hip/hip_runtime.h>
#include <hip/hip_bf16.h>

// ---------------------------------------------------------------- constants
#define BSZ  4096
#define NNODE 64
#define FEATD 128
#define OBSD 128
#define LATD 32
#define HEADS 4
#define HIDD 128   // HEADS*LATD

// d_out layout (flat f32, reference return order)
#define O_SOFT 0L
#define O_HARD ((long)BSZ * 256)
#define O_ATT  ((long)BSZ * 512)
#define O_LAT1 ((long)BSZ * 512 + (long)BSZ * 512)
#define O_LAT2 (O_LAT1 + (long)BSZ * 32)

typedef __attribute__((ext_vector_type(16))) __bf16 v16bf;
typedef __attribute__((ext_vector_type(8)))  __bf16 v8bf;
typedef __attribute__((ext_vector_type(8)))  float  v8f;
typedef __attribute__((ext_vector_type(4)))  unsigned int u32x4;
typedef __attribute__((ext_vector_type(8)))  int i32x8;
typedef __attribute__((ext_vector_type(4)))  int i32x4;

#if defined(__HIP_DEVICE_COMPILE__) && __has_builtin(__builtin_amdgcn_tensor_load_to_lds)
#define USE_TDM 1
#else
#define USE_TDM 0
#endif

// ---------------------------------------------------------------- helpers
__device__ __forceinline__ __bf16 f32_to_bf16(float f) {
  unsigned int u = __float_as_uint(f);
  unsigned int r = (u + 0x7FFFu + ((u >> 16) & 1u)) >> 16;
  unsigned short s = (unsigned short)r;
  __bf16 h;
  __builtin_memcpy(&h, &s, 2);
  return h;
}

// A fragment (16x32 bf16): lane's 16 elements are one contiguous
// 16-element run (k = 16*half + e) from a row-major bf16 LDS tile.
__device__ __forceinline__ v16bf afrag_lds(const __bf16* X, int ldk, int k0) {
  const int lane = threadIdx.x & 31;
  const int half = lane >> 4, m = lane & 15;
  return *(const v16bf*)(X + m * ldk + k0 + half * 16);
}

// GEMM tile: acc += Xs[16,K](row-major bf16 LDS) * Wp (pre-swizzled bf16
// B-fragments in global: tile (nt,kt) -> 32 lanes x 16 contiguous bf16).
template <int K>
__device__ __forceinline__ v8f gemm_tile(const __bf16* Xs,
                                         const __bf16* __restrict__ Wp,
                                         int n0) {
  constexpr int KT = K / 32;
  const int lane = threadIdx.x & 31;
  const __bf16* wbase = Wp + ((size_t)(n0 >> 4) * KT) * 512 + lane * 16;
  v8f acc = {};
#pragma unroll
  for (int kt = 0; kt < KT; ++kt) {
    if (kt + 1 < KT)
      __builtin_prefetch((const void*)(wbase + (size_t)(kt + 1) * 512), 0, 1);
    v16bf a = afrag_lds(Xs, K, kt * 32);
    v16bf b = *(const v16bf*)(wbase + (size_t)kt * 512);
    acc = __builtin_amdgcn_wmma_f32_16x16x32_bf16(false, a, false, b,
                                                  (short)0, acc, false, false);
  }
  return acc;
}

// store C/D tile (ISA f32 C/D layout) to bf16 LDS with bias (+optional ReLU)
__device__ __forceinline__ void store_bf16_tile(__bf16* Y, int ldn, int n0,
                                                v8f c, const float* bias,
                                                bool relu) {
  const int lane = threadIdx.x & 31;
  const int half = lane >> 4, n = lane & 15;
  float bb = bias ? bias[n0 + n] : 0.f;
#pragma unroll
  for (int j = 0; j < 8; ++j) {
    float x = c[j] + bb;
    if (relu) x = fmaxf(x, 0.f);
    Y[(j + half * 8) * ldn + n0 + n] = f32_to_bf16(x);
  }
}

__device__ __forceinline__ void store_f32_tile(float* Y, int ldn, int n0,
                                               v8f c, const float* bias) {
  const int lane = threadIdx.x & 31;
  const int half = lane >> 4, n = lane & 15;
  float bb = bias ? bias[n0 + n] : 0.f;
#pragma unroll
  for (int j = 0; j < 8; ++j) Y[(j + half * 8) * ldn + n0 + n] = c[j] + bb;
}

// ---------------------------------------------------------------- pack
// Convert f32 weight [K,N] (or transposed source [N,K]) into bf16 B-fragment
// order: out[(((nt*KT)+kt)*32 + lane)*16 + e] = W[kt*32 + half*16 + e][nt*16+n]
__global__ __launch_bounds__(256) void k_pack(const float* __restrict__ W,
                                              int K, int N, int transposed,
                                              __bf16* __restrict__ out) {
  size_t idx = (size_t)blockIdx.x * 256 + threadIdx.x;
  if (idx >= (size_t)K * N) return;
  int e = idx & 15;
  int lane = (idx >> 4) & 31;
  size_t tile = idx >> 9;
  int KT = K / 32;
  int kt = (int)(tile % KT);
  int nt = (int)(tile / KT);
  int k = kt * 32 + (lane >> 4) * 16 + e;
  int n = nt * 16 + (lane & 15);
  float v = transposed ? W[(size_t)n * K + k] : W[(size_t)k * N + n];
  out[idx] = f32_to_bf16(v);
}

// ---------------------------------------------------------------- kernel 1
// trans MLP (256->512->256->32) + channel-LN + q = t_ln * Wq^T -> q_ws[B,128]
__global__ __launch_bounds__(256) void k_trans(
    const float* __restrict__ c1, const float* __restrict__ c2,
    const __bf16* __restrict__ W0p, const float* __restrict__ b0,
    const __bf16* __restrict__ W1p, const float* __restrict__ b1,
    const __bf16* __restrict__ W2p, const float* __restrict__ b2,
    const __bf16* __restrict__ Wqp, const float* __restrict__ lng,
    const float* __restrict__ lnb, float* __restrict__ q_ws) {
  __shared__ __align__(32) __bf16 sX[16 * 256];
  __shared__ __align__(32) __bf16 sH1[16 * 512];
  __shared__ __align__(32) __bf16 sH2[16 * 256];
  __shared__ float  sT[16 * 32];
  __shared__ __align__(32) __bf16 sTb[16 * 32];
  __shared__ float  sMu[16], sRs[16];
  const int r0 = blockIdx.x * 16;
  const int wave = threadIdx.x >> 5;

  for (int i = threadIdx.x; i < 16 * 256; i += 256) {
    int m = i >> 8, c = i & 255;
    float v = (c < OBSD) ? c1[(r0 + m) * OBSD + c]
                         : c2[(r0 + m) * OBSD + (c - OBSD)];
    sX[i] = f32_to_bf16(v);
  }
  __syncthreads();
#pragma unroll
  for (int t = 0; t < 4; ++t) {
    int n0 = (wave * 4 + t) * 16;
    store_bf16_tile(sH1, 512, n0, gemm_tile<256>(sX, W0p, n0), b0, true);
  }
  __syncthreads();
#pragma unroll
  for (int t = 0; t < 2; ++t) {
    int n0 = (wave * 2 + t) * 16;
    store_bf16_tile(sH2, 256, n0, gemm_tile<512>(sH1, W1p, n0), b1, true);
  }
  __syncthreads();
  if (wave < 2) {
    int n0 = wave * 16;
    store_f32_tile(sT, 32, n0, gemm_tile<256>(sH2, W2p, n0), b2);
  }
  __syncthreads();
  if (threadIdx.x < 16) {
    int m = threadIdx.x;
    float mu = 0.f;
    for (int i = 0; i < 32; ++i) mu += sT[m * 32 + i];
    mu *= (1.f / 32.f);
    float var = 0.f;
    for (int i = 0; i < 32; ++i) { float d = sT[m * 32 + i] - mu; var += d * d; }
    var *= (1.f / 32.f);
    sMu[m] = mu;
    sRs[m] = rsqrtf(var + 1e-5f);
  }
  __syncthreads();
  for (int i = threadIdx.x; i < 16 * 32; i += 256) {
    int m = i >> 5, c = i & 31;
    sTb[i] = f32_to_bf16((sT[i] - sMu[m]) * sRs[m] * lng[c] + lnb[c]);
  }
  __syncthreads();
  {  // q = Tln[16,32] * Wq^T[32,128]
    int n0 = wave * 16;
    v8f acc = gemm_tile<32>(sTb, Wqp, n0);
    int lane = threadIdx.x & 31, half = lane >> 4, n = lane & 15;
#pragma unroll
    for (int j = 0; j < 8; ++j)
      q_ws[(r0 + j + half * 8) * HIDD + n0 + n] = acc[j];
  }
}

// ---------------------------------------------------------------- kernel 2
// ig MLP (128->256->256->32) + channel-LN + k' = e_ln*Wk^T + qk dots
__global__ __launch_bounds__(256) void k_ig(
    const float* __restrict__ ig, const __bf16* __restrict__ W0p,
    const float* __restrict__ b0, const __bf16* __restrict__ W1p,
    const float* __restrict__ b1, const __bf16* __restrict__ W2p,
    const float* __restrict__ b2, const __bf16* __restrict__ Wkp,
    const float* __restrict__ lng, const float* __restrict__ lnb,
    const float* __restrict__ q_ws, float* __restrict__ qk_ws) {
  __shared__ __align__(32) __bf16 sX[16 * 128];
  __shared__ __align__(16) float sXf[16 * 128];
  __shared__ __align__(32) __bf16 sH1[16 * 256];
  __shared__ __align__(32) __bf16 sH2[16 * 256];
  __shared__ float  sE[16 * 32];
  __shared__ __align__(32) __bf16 sEb[16 * 32];
  __shared__ float  sK[16 * 128];
  __shared__ float  sQ[128];
  __shared__ float  sMu[16], sRs[16];
  const int r0 = blockIdx.x * 16;          // row in [B*N]
  const int b = r0 >> 6;                   // 64 nodes per batch, tile-aligned
  const int nbase = r0 & 63;
  const int wave = threadIdx.x >> 5;

  if (threadIdx.x < 128) sQ[threadIdx.x] = q_ws[b * HIDD + threadIdx.x];
#if USE_TDM
  // Tensor Data Mover: DMA the 16x128 f32 activation tile into LDS.
  if (wave == 0) {
    u32x4 g0;
    i32x8 g1;
    i32x4 g2 = (i32x4)0, g3 = (i32x4)0;
    unsigned lds_off = (unsigned)(uintptr_t)(void*)sXf;
    unsigned long long ga =
        (unsigned long long)(uintptr_t)(ig + (size_t)r0 * FEATD);
    g0[0] = 1u;                                   // count=1, user descriptor
    g0[1] = lds_off;                              // lds_addr (bytes)
    g0[2] = (unsigned)ga;                         // global_addr[31:0]
    g0[3] = (unsigned)((ga >> 32) & 0x1FFFFFFu) | (2u << 30);  // addr hi, type=2
    g1[0] = (int)(2u << 16);                      // data_size = 4B
    g1[1] = (int)(128u << 16);                    // tensor_dim0 = 128
    g1[2] = (int)(16u << 16);                     // tensor_dim1 = 16
    g1[3] = (int)(128u << 16);                    // tile_dim0 = 128
    g1[4] = 16;                                   // tile_dim1 = 16
    g1[5] = 128;                                  // tensor_dim0_stride = 128
    g1[6] = 0;
    g1[7] = 0;
#if __clang_major__ >= 23
    i32x8 g4 = (i32x8)0;
    __builtin_amdgcn_tensor_load_to_lds(g0, g1, g2, g3, g4, 0);
#else
    __builtin_amdgcn_tensor_load_to_lds(g0, g1, g2, g3, 0);
#endif
#if __has_builtin(__builtin_amdgcn_s_wait_tensorcnt)
    __builtin_amdgcn_s_wait_tensorcnt(0);
#else
    asm volatile("s_wait_tensorcnt 0x0" ::: "memory");
#endif
  }
  __syncthreads();
  for (int i = threadIdx.x; i < 16 * FEATD; i += 256)
    sX[i] = f32_to_bf16(sXf[i]);
#else
  for (int i = threadIdx.x; i < 16 * FEATD; i += 256)
    sX[i] = f32_to_bf16(ig[(size_t)r0 * FEATD + i]);
#endif
  __syncthreads();
#pragma unroll
  for (int t = 0; t < 2; ++t) {
    int n0 = (wave * 2 + t) * 16;
    store_bf16_tile(sH1, 256, n0, gemm_tile<128>(sX, W0p, n0), b0, true);
  }
  __syncthreads();
#pragma unroll
  for (int t = 0; t < 2; ++t) {
    int n0 = (wave * 2 + t) * 16;
    store_bf16_tile(sH2, 256, n0, gemm_tile<256>(sH1, W1p, n0), b1, true);
  }
  __syncthreads();
  if (wave < 2) {
    int n0 = wave * 16;
    store_f32_tile(sE, 32, n0, gemm_tile<256>(sH2, W2p, n0), b2);
  }
  __syncthreads();
  if (threadIdx.x < 16) {
    int m = threadIdx.x;
    float mu = 0.f;
    for (int i = 0; i < 32; ++i) mu += sE[m * 32 + i];
    mu *= (1.f / 32.f);
    float var = 0.f;
    for (int i = 0; i < 32; ++i) { float d = sE[m * 32 + i] - mu; var += d * d; }
    var *= (1.f / 32.f);
    sMu[m] = mu;
    sRs[m] = rsqrtf(var + 1e-5f);
  }
  __syncthreads();
  for (int i = threadIdx.x; i < 16 * 32; i += 256) {
    int m = i >> 5, c = i & 31;
    sEb[i] = f32_to_bf16((sE[i] - sMu[m]) * sRs[m] * lng[c] + lnb[c]);
  }
  __syncthreads();
  {  // k' = Eln[16,32] * Wk^T[32,128]
    int n0 = wave * 16;
    store_f32_tile(sK, 128, n0, gemm_tile<32>(sEb, Wkp, n0), nullptr);
  }
  __syncthreads();
  if (threadIdx.x < 64) {  // qk[b,h,n] = (q_h . k'_h) / sqrt(32)
    int m = threadIdx.x >> 2, h = threadIdx.x & 3;
    float s = 0.f;
    for (int d = 0; d < 32; ++d) s += sQ[h * 32 + d] * sK[m * 128 + h * 32 + d];
    qk_ws[(b * HEADS + h) * NNODE + nbase + m] = s * 0.17677669529663687f;
  }
}

// ---------------------------------------------------------------- kernel 3
// softmax / gumbel / straight-through one-hot / attention gather
__global__ __launch_bounds__(256) void k_attn(
    const float* __restrict__ ig, const float* __restrict__ qk_ws,
    const float* __restrict__ gsoft, const float* __restrict__ ghard,
    float* __restrict__ out) {
  __shared__ float sQk[4 * 64], sS[4 * 64], sY[4 * 64], sHard[4 * 64];
  const int b = blockIdx.x;
  const int t = threadIdx.x;
  sQk[t] = qk_ws[(size_t)b * 256 + t];
  __syncthreads();
  if (t < 4) {
    int h = t;
    const float* gs = gsoft + ((size_t)b * HEADS + h) * NNODE;
    const float* gh = ghard + ((size_t)b * HEADS + h) * NNODE;
    float mx = -3.4e38f;
    for (int n = 0; n < 64; ++n) {
      float v = sQk[h * 64 + n] + gs[n];
      sS[h * 64 + n] = v;
      if (v > mx) mx = v;
    }
    float sum = 0.f;
    for (int n = 0; n < 64; ++n) {
      float e = __expf(sS[h * 64 + n] - mx);
      sS[h * 64 + n] = e;
      sum += e;
    }
    float inv = 1.f / sum;
    for (int n = 0; n < 64; ++n) sS[h * 64 + n] *= inv;
    mx = -3.4e38f;
    int amax = 0;
    for (int n = 0; n < 64; ++n) {
      float v = sQk[h * 64 + n] + gh[n];
      sY[h * 64 + n] = v;
      if (v > mx) { mx = v; amax = n; }
    }
    float s2 = 0.f;
    for (int n = 0; n < 64; ++n) {
      float e = __expf(sY[h * 64 + n] - mx);
      sY[h * 64 + n] = e;
      s2 += e;
    }
    float inv2 = 1.f / s2;
    for (int n = 0; n < 64; ++n) {
      float ys = sY[h * 64 + n] * inv2;
      float oh = (n == amax) ? 1.f : 0.f;
      float hard = (oh + ys) - ys;               // straight-through fwd value
      if (h == 0) hard = (n == 0) ? 1.f : 0.f;   // head 0 forced to node 0
      sHard[h * 64 + n] = hard;
    }
  }
  __syncthreads();
  out[O_SOFT + (size_t)b * 256 + t] = sS[t];
  out[O_HARD + (size_t)b * 256 + t] = sHard[t];
  for (int o = t; o < 512; o += 256) {  // att_flat[b, h*128+f]
    int h = o >> 7, f = o & 127;
    float s = 0.f;
    for (int n = 0; n < 64; ++n)
      s += ig[((size_t)b * NNODE + n) * FEATD + f] * sHard[h * 64 + n];
    out[O_ATT + (size_t)b * 512 + o] = s;
  }
}

// ---------------------------------------------------------------- kernel 4
// aig MLP (512->512->512->64) -> lat[:, :32], lat[:, 32:]
__global__ __launch_bounds__(256) void k_aig(
    const float* __restrict__ att, const __bf16* __restrict__ W0p,
    const float* __restrict__ b0, const __bf16* __restrict__ W1p,
    const float* __restrict__ b1, const __bf16* __restrict__ W2p,
    const float* __restrict__ b2, float* __restrict__ out) {
  __shared__ __align__(32) __bf16 sX[16 * 512];
  __shared__ __align__(32) __bf16 sH1[16 * 512];
  __shared__ __align__(32) __bf16 sH2[16 * 512];
  const int r0 = blockIdx.x * 16;
  const int wave = threadIdx.x >> 5;
  for (int i = threadIdx.x; i < 16 * 512; i += 256)
    sX[i] = f32_to_bf16(att[(size_t)r0 * 512 + i]);
  __syncthreads();
#pragma unroll
  for (int t = 0; t < 4; ++t) {
    int n0 = (wave * 4 + t) * 16;
    store_bf16_tile(sH1, 512, n0, gemm_tile<512>(sX, W0p, n0), b0, true);
  }
  __syncthreads();
#pragma unroll
  for (int t = 0; t < 4; ++t) {
    int n0 = (wave * 4 + t) * 16;
    store_bf16_tile(sH2, 512, n0, gemm_tile<512>(sH1, W1p, n0), b1, true);
  }
  __syncthreads();
  if (wave < 4) {
    int n0 = wave * 16;
    v8f acc = gemm_tile<512>(sH2, W2p, n0);
    int lane = threadIdx.x & 31, half = lane >> 4, n = lane & 15;
    int c = n0 + n;
    float bb = b2[c];
#pragma unroll
    for (int j = 0; j < 8; ++j) {
      float v = acc[j] + bb;
      int r = r0 + j + half * 8;
      if (c < 32)
        out[O_LAT1 + (size_t)r * 32 + c] = v;
      else
        out[O_LAT2 + (size_t)r * 32 + (c - 32)] = v;
    }
  }
}

// ---------------------------------------------------------------- launcher
extern "C" void kernel_launch(void* const* d_in, const int* in_sizes, int n_in,
                              void* d_out, int out_size, void* d_ws,
                              size_t ws_size, hipStream_t stream) {
  (void)in_sizes; (void)n_in; (void)out_size; (void)ws_size;
  const float* c1   = (const float*)d_in[0];
  const float* c2   = (const float*)d_in[1];
  const float* ig   = (const float*)d_in[2];
  const float* gs   = (const float*)d_in[3];
  const float* gh   = (const float*)d_in[4];
  const float* tW0  = (const float*)d_in[5];
  const float* tb0  = (const float*)d_in[6];
  const float* tW1  = (const float*)d_in[7];
  const float* tb1  = (const float*)d_in[8];
  const float* tW2  = (const float*)d_in[9];
  const float* tb2  = (const float*)d_in[10];
  const float* iW0  = (const float*)d_in[11];
  const float* ib0  = (const float*)d_in[12];
  const float* iW1  = (const float*)d_in[13];
  const float* ib1  = (const float*)d_in[14];
  const float* iW2  = (const float*)d_in[15];
  const float* ib2  = (const float*)d_in[16];
  const float* aW0  = (const float*)d_in[17];
  const float* ab0  = (const float*)d_in[18];
  const float* aW1  = (const float*)d_in[19];
  const float* ab1  = (const float*)d_in[20];
  const float* aW2  = (const float*)d_in[21];
  const float* ab2  = (const float*)d_in[22];
  const float* Wq   = (const float*)d_in[23];
  const float* Wk   = (const float*)d_in[24];
  const float* lnqg = (const float*)d_in[25];
  const float* lnqb = (const float*)d_in[26];
  const float* lnkg = (const float*)d_in[27];
  const float* lnkb = (const float*)d_in[28];

  float* out   = (float*)d_out;
  float* q_ws  = (float*)d_ws;                 // [B,128] f32
  float* qk_ws = q_ws + (size_t)BSZ * HIDD;    // [B,H,N] f32
  __bf16* pool = (__bf16*)(qk_ws + (size_t)BSZ * HEADS * NNODE);
  __bf16* tW0p = pool;              pool += 256 * 512;
  __bf16* tW1p = pool;              pool += 512 * 256;
  __bf16* tW2p = pool;              pool += 256 * 32;
  __bf16* iW0p = pool;              pool += 128 * 256;
  __bf16* iW1p = pool;              pool += 256 * 256;
  __bf16* iW2p = pool;              pool += 256 * 32;
  __bf16* aW0p = pool;              pool += 512 * 512;
  __bf16* aW1p = pool;              pool += 512 * 512;
  __bf16* aW2p = pool;              pool += 512 * 64;
  __bf16* Wqp  = pool;              pool += 32 * 128;
  __bf16* Wkp  = pool;              pool += 32 * 128;

  auto pack = [&](const float* W, int K, int N, int tr, __bf16* dst) {
    int total = K * N;
    k_pack<<<(total + 255) / 256, 256, 0, stream>>>(W, K, N, tr, dst);
  };
  pack(tW0, 256, 512, 0, tW0p);
  pack(tW1, 512, 256, 0, tW1p);
  pack(tW2, 256, 32, 0, tW2p);
  pack(iW0, 128, 256, 0, iW0p);
  pack(iW1, 256, 256, 0, iW1p);
  pack(iW2, 256, 32, 0, iW2p);
  pack(aW0, 512, 512, 0, aW0p);
  pack(aW1, 512, 512, 0, aW1p);
  pack(aW2, 512, 64, 0, aW2p);
  pack(Wq, 32, 128, 1, Wqp);   // Wq stored [128,32] -> transposed source
  pack(Wk, 32, 128, 1, Wkp);

  k_trans<<<BSZ / 16, 256, 0, stream>>>(c1, c2, tW0p, tb0, tW1p, tb1, tW2p,
                                        tb2, Wqp, lnqg, lnqb, q_ws);
  k_ig<<<(BSZ * NNODE) / 16, 256, 0, stream>>>(ig, iW0p, ib0, iW1p, ib1, iW2p,
                                               ib2, Wkp, lnkg, lnkb, q_ws,
                                               qk_ws);
  k_attn<<<BSZ, 256, 0, stream>>>(ig, qk_ws, gs, gh, out);
  k_aig<<<BSZ / 16, 256, 0, stream>>>(out + O_ATT, aW0p, ab0, aW1p, ab1, aW2p,
                                      ab2, out);
}
